// SwinTransformerBlock_45844480917817
// MI455X (gfx1250) — matrix operand order repaired and checked by reading, not compiled, not run on hardware
//
#include <hip/hip_runtime.h>
#include <hip/hip_bf16.h>
#include <math.h>

// ---------------- problem constants ----------------
#define B_        32
#define DIM_      256
#define WS_       7
#define SHIFT_    3
#define HEADS_    8
#define N_TOK     49
#define C2_       512
#define NWIN      64                  // windows per batch image (8x8)
#define NWB       (B_ * NWIN)         // 2048 total windows
#define NTOK_TOTAL (NWB * N_TOK)      // 100352 (divisible by 16)

typedef __bf16 bf16;
typedef __attribute__((ext_vector_type(8)))  bf16  v8bf;
typedef __attribute__((ext_vector_type(16))) bf16  v16bf;
typedef __attribute__((ext_vector_type(8)))  float v8f;

// ---------------- WMMA helpers (gfx1250, wave32) ----------------
static __device__ inline v16bf cat16(v8bf lo, v8bf hi) {
    v16bf r;
#pragma unroll
    for (int i = 0; i < 8; ++i) { r[i] = lo[i]; r[8 + i] = hi[i]; }
    return r;
}

// A-matrix fragment 16x32 bf16, source row-major (M, K) with leading dim ldK.
// lane%16 -> row; lane/16 selects K-halves {0..7,16..23} vs {8..15,24..31}.
static __device__ inline v16bf loadA(const bf16* base, int ldK, int m0, int kbase, int lane) {
    int half = lane >> 4;
    int r    = lane & 15;
    const bf16* p = base + (size_t)(m0 + r) * ldK + kbase + half * 8;
    v8bf lo = *(const v8bf*)p;
    v8bf hi = *(const v8bf*)(p + 16);
    return cat16(lo, hi);
}

// B-matrix fragment 32x16 bf16, source row-major (N, K) with leading dim ldK
// (i.e. B[k][n] = src[n][k]).  lane%16 -> column n; lane/16 selects K-halves {0..7,16..23} vs {8..15,24..31}.
static __device__ inline v16bf loadB(const bf16* base, int ldK, int n0, int kbase, int lane) {
    int khalf = lane >> 4;
    int c     = lane & 15;
    const bf16* p = base + (size_t)(n0 + c) * ldK + kbase + khalf * 8;
    v8bf lo = *(const v8bf*)p;
    v8bf hi = *(const v8bf*)(p + 16);
    return cat16(lo, hi);
}

static __device__ inline v8f wmma_bf16(v16bf a, v16bf b, v8f c) {
    return __builtin_amdgcn_wmma_f32_16x16x32_bf16(false, a, false, b, (short)0, c, false, false);
}

static __device__ inline float wave_sum(float v) {
#pragma unroll
    for (int o = 16; o > 0; o >>= 1) v += __shfl_xor(v, o, 32);
    return v;
}

// ---------------- kernel 0: f32 -> bf16 weight convert ----------------
__global__ __launch_bounds__(256) void f2bf_kernel(const float* __restrict__ src,
                                                   bf16* __restrict__ dst, int n) {
    int i = blockIdx.x * 256 + threadIdx.x;
    if (i < n) dst[i] = (bf16)src[i];
}

// ---------------- kernel 1: LN1 + cyclic shift + window partition + concat ----------------
// One wave per token: produces XY[token][0:256]=LN(x), [256:512]=LN(y) in bf16.
__global__ __launch_bounds__(256) void prep_kernel(const float* __restrict__ x,
                                                   const float* __restrict__ y,
                                                   const float* __restrict__ g1,
                                                   const float* __restrict__ b1,
                                                   bf16* __restrict__ XY) {
    int gtid = blockIdx.x * 256 + threadIdx.x;
    int tok  = gtid >> 5;
    int lane = threadIdx.x & 31;
    if (tok >= NTOK_TOTAL) return;
    int win = tok / N_TOK, t = tok - win * N_TOK;
    int bb = win >> 6, wIdx = win & 63;
    int wh = wIdx >> 3, wwi = wIdx & 7;
    int i = t / 7, j = t - i * 7;
    int hs = wh * 7 + i, wsc = wwi * 7 + j;
    int hsrc = hs + SHIFT_;  if (hsrc >= 56) hsrc -= 56;   // roll(-SHIFT)
    int wsrc = wsc + SHIFT_; if (wsrc >= 56) wsrc -= 56;
    size_t pix = ((size_t)bb * 3136 + hsrc * 56 + wsrc) * DIM_;
    bf16* outp = XY + (size_t)tok * C2_;
#pragma unroll
    for (int mmod = 0; mmod < 2; ++mmod) {
        const float* p = (mmod == 0 ? x : y) + pix;
        float v[8];
        float s = 0.f;
#pragma unroll
        for (int e = 0; e < 8; ++e) { int c = e * 32 + lane; v[e] = p[c]; s += v[e]; }
        s = wave_sum(s);
        float mean = s * (1.f / 256.f);
        float q = 0.f;
#pragma unroll
        for (int e = 0; e < 8; ++e) { float d = v[e] - mean; q += d * d; }
        q = wave_sum(q);
        float rstd = rsqrtf(q * (1.f / 256.f) + 1e-5f);
#pragma unroll
        for (int e = 0; e < 8; ++e) {
            int c = e * 32 + lane;
            outp[mmod * DIM_ + c] = (bf16)((v[e] - mean) * rstd * g1[c] + b1[c]);
        }
    }
}

// ---------------- kernel 2: fused windowed attention ----------------
// One block (256 threads / 8 waves) per window.  LDS (dynamic, 168 KB):
//   xyT  64x512 bf16 | outT 64x512 bf16 | qT/kT/vT 64x64 bf16 | S 64x64 f32
__global__ __launch_bounds__(256) void attn_kernel(const bf16* __restrict__ XY,
                                                   const bf16* __restrict__ Wqkv,
                                                   const float* __restrict__ bqkv,
                                                   const float* __restrict__ rpb,
                                                   const bf16* __restrict__ Wproj,
                                                   const float* __restrict__ bproj,
                                                   const float* __restrict__ xin,
                                                   float* __restrict__ out) {
    extern __shared__ char smem[];
    bf16*  xyT  = (bf16*)smem;                      // 65536 B
    bf16*  outT = (bf16*)(smem + 65536);            // 65536 B
    bf16*  qT   = (bf16*)(smem + 131072);           // 8192 B (reused as P)
    bf16*  kT   = (bf16*)(smem + 139264);           // 8192 B
    bf16*  vT   = (bf16*)(smem + 147456);           // 8192 B  (d, token) layout
    float* S    = (float*)(smem + 155648);          // 16384 B

    int tid = threadIdx.x;
    int wv = tid >> 5, lane = tid & 31;
    int win = blockIdx.x;
    int bb = win >> 6, wIdx = win & 63;
    int wh = wIdx >> 3, wwi = wIdx & 7;

    // stage window tokens into LDS, zero-pad rows 49..63
#pragma unroll
    for (int it = 0; it < 16; ++it) {
        int chunk = tid + 256 * it;          // 4096 chunks of 8 bf16
        int row = chunk >> 6;
        int co  = (chunk & 63) * 8;
        uint4 val = make_uint4(0u, 0u, 0u, 0u);
        if (row < N_TOK)
            val = *(const uint4*)(XY + ((size_t)win * N_TOK + row) * C2_ + co);
        *(uint4*)(xyT + (size_t)row * C2_ + co) = val;
    }

    const float scale = 0.17677669529663687f;  // (DIM/HEADS)^-0.5 = 32^-0.5

    for (int h = 0; h < HEADS_; ++h) {
        __syncthreads();
        // ---- QKV for this head: 48 16x16 tiles, 6 per wave, K=512 ----
#pragma unroll
        for (int jj = 0; jj < 6; ++jj) {
            int j = wv * 6 + jj;
            int which = j >> 4;           // 0=q 1=k 2=v
            int t = j & 15;
            int mt = t >> 2, nt = t & 3;
            int ncol0 = which * 512 + h * 64 + nt * 16;
            v8f acc = {0.f, 0.f, 0.f, 0.f, 0.f, 0.f, 0.f, 0.f};
            for (int kk = 0; kk < 16; ++kk) {
                v16bf a  = loadA(xyT, C2_, mt * 16, kk * 32, lane);
                v16bf bf = loadB(Wqkv, C2_, ncol0, kk * 32, lane);
                acc = wmma_bf16(a, bf, acc);
            }
            int half = lane >> 4, c = lane & 15;
            int colL = nt * 16 + c;
            float bias = bqkv[ncol0 + c];
#pragma unroll
            for (int r = 0; r < 8; ++r) {
                int row = mt * 16 + r + half * 8;
                float v = acc[r] + bias;
                if (which == 0)      qT[row * 64 + colL] = (bf16)(v * scale);
                else if (which == 1) kT[row * 64 + colL] = (bf16)v;
                else                 vT[colL * 64 + row] = (bf16)v;   // transposed
            }
        }
        __syncthreads();
        // ---- S = q @ k^T : 16 tiles, 2 per wave, K=64 ----
#pragma unroll
        for (int ss = 0; ss < 2; ++ss) {
            int t = wv * 2 + ss;
            int mt = t >> 2, nt = t & 3;
            v8f acc = {0.f, 0.f, 0.f, 0.f, 0.f, 0.f, 0.f, 0.f};
            for (int kk = 0; kk < 2; ++kk) {
                v16bf a  = loadA(qT, 64, mt * 16, kk * 32, lane);
                v16bf bf = loadB(kT, 64, nt * 16, kk * 32, lane);
                acc = wmma_bf16(a, bf, acc);
            }
            int half = lane >> 4, c = lane & 15;
#pragma unroll
            for (int r = 0; r < 8; ++r)
                S[(mt * 16 + r + half * 8) * 64 + nt * 16 + c] = acc[r];
        }
        __syncthreads();
        // ---- bias + shift-mask + softmax (row per thread), write P into qT ----
        if (tid < 64) {
            int n = tid;
            float* srow = S + n * 64;
            int reg_n = 0, rn7 = 0, cn7 = 0;
            bool nvalid = n < N_TOK;
            if (nvalid) {
                rn7 = n / 7; cn7 = n - rn7 * 7;
                int hg = wh * 7 + rn7, wg = wwi * 7 + cn7;
                int hr = (hg < 49) ? 0 : ((hg < 53) ? 1 : 2);
                int wr = (wg < 49) ? 0 : ((wg < 53) ? 1 : 2);
                reg_n = hr * 3 + wr;
            }
            float mx = -1e30f;
            for (int m = 0; m < 64; ++m) {
                float v;
                if (m < N_TOK) {
                    v = srow[m];
                    if (nvalid) {
                        int rm = m / 7, cm = m - rm * 7;
                        int idx = (rn7 - rm + 6) * 13 + (cn7 - cm + 6);
                        v += rpb[idx * HEADS_ + h];
                        int hg = wh * 7 + rm, wg = wwi * 7 + cm;
                        int hr = (hg < 49) ? 0 : ((hg < 53) ? 1 : 2);
                        int wr = (wg < 49) ? 0 : ((wg < 53) ? 1 : 2);
                        if (hr * 3 + wr != reg_n) v -= 100.f;
                    }
                } else {
                    v = -1e30f;                 // padded key columns
                }
                srow[m] = v;
                mx = fmaxf(mx, v);
            }
            float ssum = 0.f;
            for (int m = 0; m < 64; ++m) {
                float e = expf(srow[m] - mx);
                srow[m] = e;
                ssum += e;
            }
            float inv = 1.f / ssum;
            bf16* prow = qT + n * 64;           // reuse q buffer for P
            for (int m = 0; m < 64; ++m) prow[m] = (bf16)(srow[m] * inv);
        }
        __syncthreads();
        // ---- out_h = P @ V : 16 tiles, 2 per wave, K=64 (B from vT) ----
#pragma unroll
        for (int ss = 0; ss < 2; ++ss) {
            int t = wv * 2 + ss;
            int mt = t >> 2, nt = t & 3;
            v8f acc = {0.f, 0.f, 0.f, 0.f, 0.f, 0.f, 0.f, 0.f};
            for (int kk = 0; kk < 2; ++kk) {
                v16bf a  = loadA(qT, 64, mt * 16, kk * 32, lane);   // P
                v16bf bf = loadB(vT, 64, nt * 16, kk * 32, lane);
                acc = wmma_bf16(a, bf, acc);
            }
            int half = lane >> 4, c = lane & 15;
#pragma unroll
            for (int r = 0; r < 8; ++r) {
                int row = mt * 16 + r + half * 8;
                outT[(size_t)row * C2_ + h * 64 + nt * 16 + c] = (bf16)acc[r];
            }
        }
    }
    __syncthreads();
    // ---- proj (512 -> 256) + window reverse + roll(+SHIFT) + residual ----
#pragma unroll
    for (int pp = 0; pp < 8; ++pp) {
        int j = wv * 8 + pp;                 // 64 tiles: 4 M x 16 N
        int mt = j >> 4, nt = j & 15;
        v8f acc = {0.f, 0.f, 0.f, 0.f, 0.f, 0.f, 0.f, 0.f};
        for (int kk = 0; kk < 16; ++kk) {
            v16bf a  = loadA(outT, C2_, mt * 16, kk * 32, lane);
            v16bf bf = loadB(Wproj, C2_, nt * 16, kk * 32, lane);
            acc = wmma_bf16(a, bf, acc);
        }
        int half = lane >> 4, c = lane & 15;
        int col = nt * 16 + c;
        float bp = bproj[col];
#pragma unroll
        for (int r = 0; r < 8; ++r) {
            int row = mt * 16 + r + half * 8;
            if (row < N_TOK) {
                int ii = row / 7, jj2 = row - ii * 7;
                int hd = wh * 7 + ii + SHIFT_;  if (hd >= 56) hd -= 56;
                int wd = wwi * 7 + jj2 + SHIFT_; if (wd >= 56) wd -= 56;
                size_t idx = ((size_t)bb * 3136 + hd * 56 + wd) * DIM_ + col;
                out[idx] = xin[idx] + acc[r] + bp;
            }
        }
    }
}

// ---------------- kernel 3: fused LN2 + MLP + residual (in-place on d_out) ----------------
__global__ __launch_bounds__(256) void mlp_kernel(const float* __restrict__ g2,
                                                  const float* __restrict__ b2,
                                                  const bf16* __restrict__ w1,
                                                  const float* __restrict__ bb1,
                                                  const bf16* __restrict__ w2,
                                                  const float* __restrict__ bb2,
                                                  float* __restrict__ io) {
    __shared__ float xt[16 * 256];
    __shared__ bf16  hbf[16 * 256];
    __shared__ bf16  a1[16 * 1024];
    int tid = threadIdx.x;
    int wv = tid >> 5, lane = tid & 31;
    size_t row0 = (size_t)blockIdx.x * 16;

#pragma unroll
    for (int it = 0; it < 16; ++it) {
        int idx = tid + 256 * it;
        xt[idx] = io[row0 * 256 + idx];
    }
    __syncthreads();
    // LN2: 2 rows per wave
#pragma unroll
    for (int rr = 0; rr < 2; ++rr) {
        int r = wv * 2 + rr;
        float v[8];
        float s = 0.f;
#pragma unroll
        for (int e = 0; e < 8; ++e) { int c = e * 32 + lane; v[e] = xt[r * 256 + c]; s += v[e]; }
        s = wave_sum(s);
        float mean = s * (1.f / 256.f);
        float q = 0.f;
#pragma unroll
        for (int e = 0; e < 8; ++e) { float d = v[e] - mean; q += d * d; }
        q = wave_sum(q);
        float rstd = rsqrtf(q * (1.f / 256.f) + 1e-5f);
#pragma unroll
        for (int e = 0; e < 8; ++e) {
            int c = e * 32 + lane;
            hbf[r * 256 + c] = (bf16)((v[e] - mean) * rstd * g2[c] + b2[c]);
        }
    }
    __syncthreads();
    // fc1 (K=256, N=1024) + exact GELU
#pragma unroll
    for (int p = 0; p < 8; ++p) {
        int nt = wv * 8 + p;
        v8f acc = {0.f, 0.f, 0.f, 0.f, 0.f, 0.f, 0.f, 0.f};
        for (int kk = 0; kk < 8; ++kk) {
            v16bf a  = loadA(hbf, 256, 0, kk * 32, lane);
            v16bf bf = loadB(w1, 256, nt * 16, kk * 32, lane);
            acc = wmma_bf16(a, bf, acc);
        }
        int half = lane >> 4, c = lane & 15;
        int col = nt * 16 + c;
        float b1v = bb1[col];
#pragma unroll
        for (int r = 0; r < 8; ++r) {
            int row = r + half * 8;
            float u = acc[r] + b1v;
            u = 0.5f * u * (1.f + erff(u * 0.70710678118654752f));
            a1[row * 1024 + col] = (bf16)u;
        }
    }
    __syncthreads();
    // fc2 (K=1024, N=256) + residual, write back in place
#pragma unroll
    for (int p = 0; p < 2; ++p) {
        int nt = wv * 2 + p;
        v8f acc = {0.f, 0.f, 0.f, 0.f, 0.f, 0.f, 0.f, 0.f};
        for (int kk = 0; kk < 32; ++kk) {
            v16bf a  = loadA(a1, 1024, 0, kk * 32, lane);
            v16bf bf = loadB(w2, 1024, nt * 16, kk * 32, lane);
            acc = wmma_bf16(a, bf, acc);
        }
        int half = lane >> 4, c = lane & 15;
        int col = nt * 16 + c;
        float b2v = bb2[col];
#pragma unroll
        for (int r = 0; r < 8; ++r) {
            int row = r + half * 8;
            io[(row0 + row) * 256 + col] = xt[row * 256 + col] + acc[r] + b2v;
        }
    }
}

// ---------------- host launcher ----------------
extern "C" void kernel_launch(void* const* d_in, const int* in_sizes, int n_in,
                              void* d_out, int out_size, void* d_ws, size_t ws_size,
                              hipStream_t stream) {
    (void)in_sizes; (void)n_in; (void)out_size; (void)ws_size;
    const float* x      = (const float*)d_in[0];
    const float* y      = (const float*)d_in[1];
    const float* g1     = (const float*)d_in[2];
    const float* b1     = (const float*)d_in[3];
    const float* rpb    = (const float*)d_in[4];
    const float* w_qkv  = (const float*)d_in[5];
    const float* b_qkv  = (const float*)d_in[6];
    const float* w_proj = (const float*)d_in[7];
    const float* b_proj = (const float*)d_in[8];
    const float* g2     = (const float*)d_in[9];
    const float* b2     = (const float*)d_in[10];
    const float* w_fc1  = (const float*)d_in[11];
    const float* b_fc1  = (const float*)d_in[12];
    const float* w_fc2  = (const float*)d_in[13];
    const float* b_fc2  = (const float*)d_in[14];

    char* ws = (char*)d_ws;
    bf16* XY = (bf16*)ws;
    size_t off = (size_t)NTOK_TOTAL * C2_ * sizeof(bf16);
    bf16* wqkvbf  = (bf16*)(ws + off); off += (size_t)1536 * 512 * sizeof(bf16);
    bf16* wprojbf = (bf16*)(ws + off); off += (size_t)256 * 512 * sizeof(bf16);
    bf16* wfc1bf  = (bf16*)(ws + off); off += (size_t)1024 * 256 * sizeof(bf16);
    bf16* wfc2bf  = (bf16*)(ws + off); off += (size_t)256 * 1024 * sizeof(bf16);

    f2bf_kernel<<<(1536 * 512 + 255) / 256, 256, 0, stream>>>(w_qkv, wqkvbf, 1536 * 512);
    f2bf_kernel<<<(256 * 512 + 255) / 256, 256, 0, stream>>>(w_proj, wprojbf, 256 * 512);
    f2bf_kernel<<<(1024 * 256 + 255) / 256, 256, 0, stream>>>(w_fc1, wfc1bf, 1024 * 256);
    f2bf_kernel<<<(256 * 1024 + 255) / 256, 256, 0, stream>>>(w_fc2, wfc2bf, 256 * 1024);

    prep_kernel<<<NTOK_TOTAL / 8, 256, 0, stream>>>(x, y, g1, b1, XY);

    const size_t ATTN_LDS = 172032;  // 168 KB dynamic LDS (<= 320 KB/WGP)
    (void)hipFuncSetAttribute((const void*)attn_kernel,
                              hipFuncAttributeMaxDynamicSharedMemorySize, (int)ATTN_LDS);
    attn_kernel<<<NWB, 256, ATTN_LDS, stream>>>(XY, wqkvbf, b_qkv, rpb, wprojbf, b_proj,
                                                x, (float*)d_out);

    mlp_kernel<<<NTOK_TOTAL / 16, 256, 0, stream>>>(g2, b2, wfc1bf, b_fc1, wfc2bf, b_fc2,
                                                    (float*)d_out);
}
